// FunctionalResiduePredUnit_76622216561173
// MI455X (gfx1250) — matrix operand order repaired and checked by reading, not compiled, not run on hardware
//
#include <hip/hip_runtime.h>
#include <hip/hip_bf16.h>
#include <stdint.h>

// ---------------------------------------------------------------------------
// FunctionalResiduePredUnit for MI455X (gfx1250, wave32, WMMA)
//   - 2-hop masks as dense f16 WMMA matmuls (exact for 0/1 data, f32 accum),
//     double-buffered LDS staging via gfx1250 async global->LDS b128 copies
//     (ASYNCcnt pipelining: issue next stage, s_wait_asynccnt 2, barrier)
//   - h = x_orig @ W2^T as a WMMA GEMV (f32 -> f16 convert, branchless B)
//   - per-ego GAT + softmax via CSR-by-dst gather (deterministic build)
// ---------------------------------------------------------------------------

typedef _Float16 half8  __attribute__((ext_vector_type(8)));
typedef _Float16 half16 __attribute__((ext_vector_type(16)));
typedef float    float8 __attribute__((ext_vector_type(8)));

#define NN   2048
#define EE   16384
#define DIN  1280
#define NEGV (-1e30f)

// ---- gfx1250 async global->LDS copies (ISA §15.18 VGLOBAL ops 95-98) -------
// VDST = LDS byte address in a VGPR (low 32 bits of the flat shared pointer,
// per ISA §10.2), VADDR = 64-bit global address,
// SADDR = off. Tracked with ASYNCcnt; async loads complete in order.
__device__ __forceinline__ void async_copy_b128(const _Float16* g, _Float16* l) {
  uint32_t lds = (uint32_t)(uintptr_t)l;
  asm volatile("global_load_async_to_lds_b128 %0, %1, off"
               :: "v"(lds), "v"(g) : "memory");
}
__device__ __forceinline__ void wait_async0() {
  asm volatile("s_wait_asynccnt 0x0" ::: "memory");
}
__device__ __forceinline__ void wait_async2() {   // allow 2 newest in flight
  asm volatile("s_wait_asynccnt 0x2" ::: "memory");
}

// ---------------- init: zero ADJ, M = I, zero CSR counts --------------------
__global__ void k_init(_Float16* __restrict__ Mcur, _Float16* __restrict__ adj,
                       int* __restrict__ counts) {
  int idx = blockIdx.x * blockDim.x + threadIdx.x;
  if (idx < NN * NN) {
    adj[idx] = (_Float16)0.f;
    int r = idx >> 11, c = idx & (NN - 1);
    Mcur[idx] = (r == c) ? (_Float16)1.f : (_Float16)0.f;
  }
  if (idx < NN) counts[idx] = 0;
}

// ---------------- adjacency (transposed layout ADJ[s][d]) + dst histogram ---
__global__ void k_adj(const int* __restrict__ ei, _Float16* __restrict__ adj,
                      int* __restrict__ counts) {
  int e = blockIdx.x * blockDim.x + threadIdx.x;
  if (e < EE) {
    int s = ei[e];
    int d = ei[EE + e];
    adj[(size_t)s * NN + d] = (_Float16)1.f;   // duplicates are fine (still 1)
    atomicAdd(&counts[d], 1);
  }
}

// ---------------- h[i] = sum_k x_orig[i,k] * W2[k]  (WMMA GEMV) -------------
// One wave computes a 16-row tile; only output column 0 is meaningful.
// B fragment is built branchlessly: every lane loads W2, lanes with n!=0
// multiply by zero (avoids per-element exec-mask branches).
__global__ void k_gemv(const float* __restrict__ xo, const float* __restrict__ W2,
                       float* __restrict__ h) {
  int lane  = threadIdx.x;                             // 0..31
  int tile  = blockIdx.x * blockDim.y + threadIdx.y;   // 0..127
  int m     = lane & 15;
  int koff  = (lane < 16) ? 0 : 8;                     // A-fragment K offset
  int khalf = (lane < 16) ? 0 : 16;                    // unused (B also uses koff)
  int n     = lane & 15;
  _Float16 sel = (n == 0) ? (_Float16)1.f : (_Float16)0.f;

  const float* arow = xo + (size_t)(tile * 16 + m) * DIN;
  float8 c = {};
  for (int kb = 0; kb < DIN; kb += 32) {
    half16 a, b;
#pragma unroll
    for (int e = 0; e < 8; ++e) a[e]     = (_Float16)arow[kb + koff + e];
#pragma unroll
    for (int e = 0; e < 8; ++e) a[e + 8] = (_Float16)arow[kb + koff + 16 + e];
#pragma unroll
    for (int e = 0; e < 16; ++e)
      b[e] = sel * (_Float16)W2[kb + koff + (e & 7) + ((e >> 3) << 4)];
    c = __builtin_amdgcn_wmma_f32_16x16x32_f16(false, a, false, b,
                                               (short)0, c, false, false);
  }
  if (n == 0) {
    int mbase = (lane < 16) ? 0 : 8;
#pragma unroll
    for (int r = 0; r < 8; ++r) h[tile * 16 + mbase + r] = c[r];
  }
}

// ---------------- self-loop softmax weights ---------------------------------
__global__ void k_selfw(const float* __restrict__ h, const float* __restrict__ as,
                        const float* __restrict__ ad, float* __restrict__ selfw) {
  int i = blockIdx.x * blockDim.x + threadIdx.x;
  if (i < NN) {
    float z = (as[0] + ad[0]) * h[i];
    z = z > 0.f ? z : 0.2f * z;                  // leaky relu, slope 0.2
    selfw[i] = __expf(z);                        // global shift 0 (logits O(0.2))
  }
}

// ---------------- exclusive scan of dst histogram (serial, tiny) ------------
__global__ void k_scan(const int* __restrict__ counts, int* __restrict__ offs) {
  if (blockIdx.x == 0 && threadIdx.x == 0) {
    int acc = 0;
    for (int i = 0; i < NN; ++i) { offs[i] = acc; acc += counts[i]; }
    offs[NN] = acc;
  }
}

// ---------------- deterministic CSR bucket fill (one thread per dst) --------
__global__ void k_scatter(const int* __restrict__ ei, const float* __restrict__ h,
                          const float* __restrict__ asp, const float* __restrict__ adp,
                          const int* __restrict__ offs, int* __restrict__ srcs,
                          float* __restrict__ pexp, float* __restrict__ ph) {
  int i = blockIdx.x * blockDim.x + threadIdx.x;
  if (i < NN) {
    int pos  = offs[i];
    float ac = asp[0];
    float hd = adp[0] * h[i];
    for (int e = 0; e < EE; ++e) {
      if (ei[EE + e] == i) {
        int s   = ei[e];
        float z = ac * h[s] + hd;
        z = z > 0.f ? z : 0.2f * z;
        float w = __expf(z);
        srcs[pos] = s; pexp[pos] = w; ph[pos] = w * h[s]; ++pos;
      }
    }
  }
}

// ---------------- one k-hop expansion: Mnew = Mold | (Mold . ADJ^T > 0) -----
// 512-thread block (16 waves in a 4x4 grid) computes a 64x64 output tile.
// K-step 64: each stage DMAs A(64x64) and B(64x64) f16 panels into LDS with
// one b128 async copy per thread per panel, double-buffered so the next
// stage's DMA overlaps the current stage's two WMMAs per wave.
__global__ void __launch_bounds__(512)
k_khop(const _Float16* __restrict__ Mold, const _Float16* __restrict__ adj,
       _Float16* __restrict__ Mnew) {
  __shared__ __align__(16) _Float16 sA[2][64 * 64];   // [buf][row v][k]
  __shared__ __align__(16) _Float16 sB[2][64 * 64];   // [buf][row s][k]

  int lane = threadIdx.x;                 // 0..31
  int w    = threadIdx.y;                 // 0..15 (wave id)
  int wy   = w >> 2, wx = w & 3;          // 4x4 wave grid
  int tid  = w * 32 + lane;               // 0..511

  int tmBase = blockIdx.y * 64;           // v rows
  int tnBase = blockIdx.x * 64;           // s rows (ADJ is [s][d])

  int m     = lane & 15;
  int n     = lane & 15;
  int koff  = (lane < 16) ? 0 : 8;
  int khalf = (lane < 16) ? 0 : 16;

  // staging assignment: 512 threads x 8 halves (b128) covers a 64x64 panel
  int srow = tid >> 3;                    // 0..63
  int scol = (tid & 7) * 8;               // 0,8,...,56

  const _Float16* gA = Mold + (size_t)(tmBase + srow) * NN + scol;
  const _Float16* gB = adj  + (size_t)(tnBase + srow) * NN + scol;
  int lofs = srow * 64 + scol;

  // prologue: stage 0 into buffer 0
  async_copy_b128(gA, &sA[0][lofs]);
  async_copy_b128(gB, &sB[0][lofs]);

  float8 c = {};
  for (int kb = 0; kb < NN; kb += 64) {
    int cur = (kb >> 6) & 1;
    if (kb + 64 < NN) {
      // prefetch next stage into the other buffer (its readers barriered out
      // at the end of the previous iteration), then wait only for the
      // current stage's two copies (async loads complete in order).
      async_copy_b128(gA + kb + 64, &sA[cur ^ 1][lofs]);
      async_copy_b128(gB + kb + 64, &sB[cur ^ 1][lofs]);
      wait_async2();
    } else {
      wait_async0();
    }
    __syncthreads();                      // everyone's current copies landed

    const _Float16* aRow = &sA[cur][(wy * 16 + m) * 64];
    const _Float16* bRow = &sB[cur][(wx * 16 + n) * 64];
#pragma unroll
    for (int ks = 0; ks < 64; ks += 32) {
      half8  alo = *(const half8*)(aRow + ks + khalf);
      half8  ahi = *(const half8*)(aRow + ks + khalf + 8);
      half16 b   = *(const half16*)(bRow + ks + khalf);
      half16 a;
#pragma unroll
      for (int e = 0; e < 8; ++e) { a[e] = alo[e]; a[e + 8] = ahi[e]; }
      c = __builtin_amdgcn_wmma_f32_16x16x32_f16(false, a, false, b,
                                                 (short)0, c, false, false);
    }
    __syncthreads();                      // protect buffers before restage
  }

  int mbase = (lane < 16) ? 0 : 8;
#pragma unroll
  for (int r = 0; r < 8; ++r) {
    size_t v = (size_t)(tmBase + wy * 16 + mbase + r);
    size_t s = (size_t)(tnBase + wx * 16 + n);
    float old = (float)Mold[v * NN + s];
    Mnew[v * NN + s] = (c[r] + old) > 0.5f ? (_Float16)1.f : (_Float16)0.f;
  }
}

// ---------------- per-ego GAT + softmax -------------------------------------
__global__ void k_scores(const _Float16* __restrict__ M, const float* __restrict__ h,
                         const float* __restrict__ selfw, const int* __restrict__ offs,
                         const int* __restrict__ srcs, const float* __restrict__ pexp,
                         const float* __restrict__ ph, const float* __restrict__ bias,
                         float* __restrict__ out) {
  __shared__ _Float16 msk[NN];
  __shared__ float    gat[NN];
  __shared__ float    red[256];

  int v = blockIdx.x;
  int t = threadIdx.x;          // 256 threads = 8 waves

  for (int i = t; i < NN; i += 256) msk[i] = M[(size_t)v * NN + i];
  __syncthreads();

  float b0 = bias[0];
  for (int i = t; i < NN; i += 256) {
    if ((float)msk[i] > 0.f) {
      float denom = selfw[i];
      float num   = selfw[i] * h[i];
      int e0 = offs[i], e1 = offs[i + 1];
      for (int e = e0; e < e1; ++e) {
        int s = srcs[e];
        if ((float)msk[s] > 0.f) { denom += pexp[e]; num += ph[e]; }
      }
      gat[i] = num / denom + b0;
    } else {
      gat[i] = NEGV;
    }
  }
  __syncthreads();

  // row max (ego always contains itself, so a valid max exists)
  float lm = NEGV;
  for (int i = t; i < NN; i += 256) lm = fmaxf(lm, gat[i]);
  red[t] = lm; __syncthreads();
  for (int s = 128; s > 0; s >>= 1) { if (t < s) red[t] = fmaxf(red[t], red[t + s]); __syncthreads(); }
  float mx = red[0]; __syncthreads();

  float ls = 0.f;
  for (int i = t; i < NN; i += 256)
    if ((float)msk[i] > 0.f) ls += __expf(gat[i] - mx);
  red[t] = ls; __syncthreads();
  for (int s = 128; s > 0; s >>= 1) { if (t < s) red[t] += red[t + s]; __syncthreads(); }
  float inv = 1.f / red[0];

  float* out_scores = out;
  float* out_mask   = out + (size_t)NN * NN;
  for (int i = t; i < NN; i += 256) {
    bool in = (float)msk[i] > 0.f;
    out_scores[(size_t)v * NN + i] = in ? __expf(gat[i] - mx) * inv : 0.f;
    out_mask  [(size_t)v * NN + i] = in ? 1.f : 0.f;
  }
}

// ---------------------------------------------------------------------------
extern "C" void kernel_launch(void* const* d_in, const int* in_sizes, int n_in,
                              void* d_out, int out_size, void* d_ws, size_t ws_size,
                              hipStream_t stream) {
  (void)in_sizes; (void)n_in; (void)out_size; (void)ws_size;
  const float* x_orig     = (const float*)d_in[1];
  const int*   edge_index = (const int*)d_in[2];
  const float* W2         = (const float*)d_in[4];
  const float* att_src    = (const float*)d_in[5];
  const float* att_dst    = (const float*)d_in[6];
  const float* bias       = (const float*)d_in[7];

  char* ws = (char*)d_ws;
  _Float16* Mcur = (_Float16*)(ws);
  _Float16* Mtmp = (_Float16*)(ws + (size_t)8  * 1024 * 1024);
  _Float16* adj  = (_Float16*)(ws + (size_t)16 * 1024 * 1024);
  float* h     = (float*)(ws + (size_t)24 * 1024 * 1024);
  float* selfw = h + NN;
  float* pexp  = selfw + NN;
  float* ph    = pexp + EE;
  int*   srcs  = (int*)(ph + EE);
  int*   offs  = srcs + EE;
  int*   cnts  = offs + (NN + 1);

  k_init   <<<(NN * NN + 255) / 256, 256, 0, stream>>>(Mcur, adj, cnts);
  k_adj    <<<(EE + 255) / 256, 256, 0, stream>>>(edge_index, adj, cnts);
  k_gemv   <<<NN / 16 / 4, dim3(32, 4), 0, stream>>>(x_orig, W2, h);
  k_selfw  <<<(NN + 255) / 256, 256, 0, stream>>>(h, att_src, att_dst, selfw);
  k_scan   <<<1, 1, 0, stream>>>(cnts, offs);
  k_scatter<<<(NN + 255) / 256, 256, 0, stream>>>(edge_index, h, att_src, att_dst,
                                                  offs, srcs, pexp, ph);
  dim3 gk(NN / 64, NN / 64);               // (32, 32) 64x64 tiles
  k_khop<<<gk, dim3(32, 16), 0, stream>>>(Mcur, adj, Mtmp);   // hop 1
  k_khop<<<gk, dim3(32, 16), 0, stream>>>(Mtmp, adj, Mcur);   // hop 2
  k_scores<<<NN, 256, 0, stream>>>(Mcur, h, selfw, offs, srcs, pexp, ph, bias,
                                   (float*)d_out);
}
